// AggregationAttentionBlock_73744588472515
// MI455X (gfx1250) — matrix operand order repaired and checked by reading, not compiled, not run on hardware
//
#include <hip/hip_runtime.h>
#include <hip/hip_bf16.h>
#include <math.h>
#include <stdint.h>

typedef _Float16 half_t;
typedef __attribute__((ext_vector_type(16))) _Float16 v16h;
typedef __attribute__((ext_vector_type(8)))  _Float16 v8h;
typedef __attribute__((ext_vector_type(8)))  float    v8f;
typedef __attribute__((ext_vector_type(4)))  int      v4i;

#define WMMA_F32_F16(A_, B_, C_) \
  __builtin_amdgcn_wmma_f32_16x16x32_f16(false, (A_), false, (B_), (short)0, (C_), false, false)

// ---------------------------------------------------------------------------
// Async global->LDS copy (CDNA5 GLOBAL_LOAD_ASYNC_TO_LDS_B128, ASYNCcnt) with
// a portable fallback. Builtin signature probed via compile diagnostics:
//   (int4 addrspace(1)* src_global, int4 addrspace(3)* dst_lds, imm, imm)
// ---------------------------------------------------------------------------
#if defined(__has_builtin)
#  if __has_builtin(__builtin_amdgcn_global_load_async_to_lds_b128)
#    define HAVE_ASYNC_LDS 1
#  endif
#endif
#ifndef HAVE_ASYNC_LDS
#  define HAVE_ASYNC_LDS 0
#endif

static __device__ __forceinline__ void async_copy16(half_t* lds, const half_t* g) {
#if HAVE_ASYNC_LDS
  __builtin_amdgcn_global_load_async_to_lds_b128(
      (__attribute__((address_space(1))) v4i*)(uintptr_t)g,
      (__attribute__((address_space(3))) v4i*)(uint32_t)(uintptr_t)lds,
      0, 0);
#else
  *(ulonglong2*)lds = *(const ulonglong2*)g;
#endif
}
static __device__ __forceinline__ void wait_async_pending2() {
#if HAVE_ASYNC_LDS
  asm volatile("s_wait_asynccnt 0x2" ::: "memory");
#endif
}
static __device__ __forceinline__ void wait_async_all() {
#if HAVE_ASYNC_LDS
  asm volatile("s_wait_asynccnt 0x0" ::: "memory");
#endif
}

static __device__ __forceinline__ v16h pack16(v8h lo, v8h hi) {
  v16h f;
#pragma unroll
  for (int i = 0; i < 8; ++i) { f[i] = lo[i]; f[i + 8] = hi[i]; }
  return f;
}

// ---------------------------------------------------------------------------
// A-fragment loader used by attention (16-bit A 16x32, contiguous k order, same as B):
// lane l: row m=l&15, k-base=16*(l>>4); elems 0..15 -> kb..kb+15
// ---------------------------------------------------------------------------
static __device__ __forceinline__ v16h load_a_frag(const half_t* __restrict__ base,
                                                   int ld, int lane16, int lh) {
  const half_t* p = base + (size_t)lane16 * ld + lh * 16;
  return pack16(*(const v8h*)p, *(const v8h*)(p + 8));
}

// ---------------------------------------------------------------------------
// WMMA GEMM tile, software-pipelined, K is compile-time so all row strides
// (16*K halves) become immediate offsets off TWO base pointers -> no 64-bit
// address arithmetic or spills in the hot loop.
// wave = 32(M) x 64(N); block = 8 waves = 128x128 tile.
// A [M,K] f16 row-major, Wt [N,K] f16 (pre-transposed).
// ---------------------------------------------------------------------------
template <int K, class Epi>
static __device__ __forceinline__ void gemm_tile(const half_t* __restrict__ A,
                                                 const half_t* __restrict__ Wt,
                                                 Epi epi) {
  const int l      = threadIdx.x & 31;
  const int wave   = threadIdx.x >> 5;
  const int lane16 = l & 15;
  const int lh     = l >> 4;
  const int m0 = blockIdx.y * 128 + (wave >> 1) * 32;
  const int n0 = blockIdx.x * 128 + (wave & 1) * 64;

  // one base pointer per lane for A and for B; all row offsets are constants
  const half_t* Ab = A  + (size_t)(m0 + lane16) * K + lh * 16;
  const half_t* Bb = Wt + (size_t)(n0 + lane16) * K + lh * 16;

  v8f z = {};
  v8f acc[2][4];
#pragma unroll
  for (int r = 0; r < 2; ++r)
#pragma unroll
    for (int t = 0; t < 4; ++t) acc[r][t] = z;

  auto lda = [&](int r, int k) -> v16h {
    const half_t* p = Ab + (size_t)r * 16 * K + k;
    return pack16(*(const v8h*)p, *(const v8h*)(p + 8));
  };
  auto ldb = [&](int t, int k) -> v16h {
    return *(const v16h*)(Bb + (size_t)t * 16 * K + k);
  };

  // prologue: fragments for k=0
  v16h a0 = lda(0, 0), a1 = lda(1, 0);
  v16h bf[4];
#pragma unroll
  for (int t = 0; t < 4; ++t) bf[t] = ldb(t, 0);

#pragma unroll 4
  for (int k0 = 0; k0 < K; k0 += 32) {
    const int kn = (k0 + 32 < K) ? (k0 + 32) : 0;  // tail reload is harmless
    v16h na0 = lda(0, kn), na1 = lda(1, kn);
    v16h nbf[4];
#pragma unroll
    for (int t = 0; t < 4; ++t) nbf[t] = ldb(t, kn);

    if (k0 + 512 < K) {  // deep prefetch into cache hierarchy
      __builtin_prefetch(Ab + k0 + 512, 0, 3);
      __builtin_prefetch(Bb + k0 + 512, 0, 3);
    }

#pragma unroll
    for (int t = 0; t < 4; ++t) {
      acc[0][t] = WMMA_F32_F16(a0, bf[t], acc[0][t]);
      acc[1][t] = WMMA_F32_F16(a1, bf[t], acc[1][t]);
    }
    a0 = na0; a1 = na1;
#pragma unroll
    for (int t = 0; t < 4; ++t) bf[t] = nbf[t];
  }
  // C layout: lane l, vgpr j -> row m0+16r+8*(l>>4)+j, col n0+16t+(l&15)
#pragma unroll
  for (int r = 0; r < 2; ++r)
#pragma unroll
    for (int t = 0; t < 4; ++t)
#pragma unroll
      for (int j = 0; j < 8; ++j)
        epi(m0 + r * 16 + lh * 8 + j, n0 + t * 16 + lane16, (float)acc[r][t][j]);
}

// ---------------------------------------------------------------------------
// fp32 [K,N] -> f16 transposed [N,K]
// ---------------------------------------------------------------------------
__global__ void k_trans(const float* __restrict__ in, half_t* __restrict__ out,
                        int K, int N) {
  int idx = blockIdx.x * 256 + threadIdx.x;
  if (idx < K * N) {
    int k = idx / N, n = idx - k * N;
    out[(size_t)n * K + k] = (half_t)in[idx];
  }
}

// ---------------------------------------------------------------------------
// block-wide sum of two values (256 threads = 8 waves)
// ---------------------------------------------------------------------------
static __device__ __forceinline__ float2 block_sum2(float s1, float s2) {
#pragma unroll
  for (int d = 16; d >= 1; d >>= 1) {
    s1 += __shfl_xor(s1, d, 32);
    s2 += __shfl_xor(s2, d, 32);
  }
  __shared__ float sh[2][8];
  int w = threadIdx.x >> 5;
  __syncthreads();
  if ((threadIdx.x & 31) == 0) { sh[0][w] = s1; sh[1][w] = s2; }
  __syncthreads();
  float t1 = 0.f, t2 = 0.f;
#pragma unroll
  for (int i = 0; i < 8; ++i) { t1 += sh[0][i]; t2 += sh[1][i]; }
  return make_float2(t1, t2);
}

// ---------------------------------------------------------------------------
// Prep: comb = (f + r)/sqrt(2) (fp32 residual), cx = LN(comb), ae = LN(all_embed)
// ---------------------------------------------------------------------------
__global__ void k_prep(const float* __restrict__ fwd, const float* __restrict__ bwd,
                       const float* __restrict__ g, const float* __restrict__ be,
                       float* __restrict__ comb, half_t* __restrict__ cx,
                       half_t* __restrict__ ae) {
  const int row = blockIdx.x;
  const int c0  = threadIdx.x * 4;
  float v[4];
  if (row < 2048) {
    int b = row >> 10, s = row & 1023;
    const float* pf = fwd + ((size_t)b * 1025 + s) * 1024 + c0;
    const float* pr = bwd + ((size_t)b * 1025 + s + 1) * 1024 + c0;
#pragma unroll
    for (int i = 0; i < 4; ++i) v[i] = (pf[i] + pr[i]) * 0.70710678118654752f;
    float* pc = comb + (size_t)row * 1024 + c0;
#pragma unroll
    for (int i = 0; i < 4; ++i) pc[i] = v[i];
    float s1 = 0.f, s2 = 0.f;
#pragma unroll
    for (int i = 0; i < 4; ++i) { s1 += v[i]; s2 += v[i] * v[i]; }
    float2 r2 = block_sum2(s1, s2);
    float mean = r2.x * (1.0f / 1024.0f);
    float var  = r2.y * (1.0f / 1024.0f) - mean * mean;
    float rstd = rsqrtf(var + 1e-5f);
    half_t* po = cx + (size_t)row * 1024 + c0;
#pragma unroll
    for (int i = 0; i < 4; ++i)
      po[i] = (half_t)((v[i] - mean) * rstd * g[c0 + i] + be[c0 + i]);
  } else {
    int rr = row - 2048;
    int b = rr >> 11, j = rr & 2047;
    const float* ps = (j < 1024)
        ? fwd + ((size_t)b * 1025 + j) * 1024 + c0
        : bwd + ((size_t)b * 1025 + (j - 1024) + 1) * 1024 + c0;
#pragma unroll
    for (int i = 0; i < 4; ++i) v[i] = ps[i];
    float s1 = 0.f, s2 = 0.f;
#pragma unroll
    for (int i = 0; i < 4; ++i) { s1 += v[i]; s2 += v[i] * v[i]; }
    float2 r2 = block_sum2(s1, s2);
    float mean = r2.x * (1.0f / 1024.0f);
    float var  = r2.y * (1.0f / 1024.0f) - mean * mean;
    float rstd = rsqrtf(var + 1e-5f);
    half_t* po = ae + (size_t)rr * 1024 + c0;
#pragma unroll
    for (int i = 0; i < 4; ++i)
      po[i] = (half_t)((v[i] - mean) * rstd * g[c0 + i] + be[c0 + i]);
  }
}

// ---------------------------------------------------------------------------
// QKV projection kernels
// ---------------------------------------------------------------------------
__global__ void k_gemm_q(const half_t* __restrict__ A, const half_t* __restrict__ Wt,
                         const float* __restrict__ bias, const float* __restrict__ pos,
                         half_t* __restrict__ qb) {
  gemm_tile<1024>(A, Wt, [&](int m, int n, float v) {
    int b = m >> 10, s = m & 1023;
    int h = n >> 6, hd = n & 63;
    v = (v + bias[n] + pos[s * 64 + hd]) * 0.125f;  // 1/sqrt(64)
    qb[(((size_t)(b * 16 + h) * 1024) + s) * 64 + hd] = (half_t)v;
  });
}

__global__ void k_gemm_k(const half_t* __restrict__ A, const half_t* __restrict__ Wt,
                         const float* __restrict__ bias, const float* __restrict__ pos,
                         half_t* __restrict__ kbuf) {
  gemm_tile<1024>(A, Wt, [&](int m, int n, float v) {
    int b = m >> 11, j = m & 2047;
    int h = n >> 6, hd = n & 63;
    v = v + bias[n] + pos[(j & 1023) * 64 + hd];    // pos2[j] = pos[j % S]
    kbuf[(((size_t)(b * 16 + h) * 2048) + j) * 64 + hd] = (half_t)v;
  });
}

__global__ void k_gemm_v(const half_t* __restrict__ A, const half_t* __restrict__ Wt,
                         const float* __restrict__ bias, half_t* __restrict__ vtb) {
  gemm_tile<1024>(A, Wt, [&](int m, int n, float v) {
    int b = m >> 11, j = m & 2047;
    int h = n >> 6, hd = n & 63;
    // transposed store [B,H,HD,2S] -> contiguous B-fragments for P@V
    vtb[(((size_t)(b * 16 + h) * 64) + hd) * 2048 + j] = (half_t)(v + bias[n]);
  });
}

// ---------------------------------------------------------------------------
// Flash attention with async-LDS double-buffered K/V tiles.
// Block = 8 waves, one (b,h), 128 query rows; wave = 16 query rows.
// qb [B,H,S,64], kbuf [B,H,2S,64], vtb [B,H,64,2S] (f16) -> ob [B,S,C] f16
// ---------------------------------------------------------------------------
__global__ void k_attn(const half_t* __restrict__ qb, const half_t* __restrict__ kbuf,
                       const half_t* __restrict__ vtb, half_t* __restrict__ ob) {
  __shared__ __align__(32) half_t psh[8][16 * 32];
  __shared__ __align__(32) half_t Ks[2][32 * 64];   // [key][hd]
  __shared__ __align__(32) half_t Vs[2][64 * 32];   // [hd][key]

  const int tid = threadIdx.x;
  const int l = tid & 31;
  const int w = tid >> 5;
  const int lane16 = l & 15, lh = l >> 4;
  const int qblk = blockIdx.x & 7;
  const int bh   = blockIdx.x >> 3;    // b*16 + h
  const int b = bh >> 4, h = bh & 15;
  const int q0 = qblk * 128 + w * 16;

  const half_t* Kbase = kbuf + (size_t)bh * 2048 * 64;
  const half_t* Vbase = vtb  + (size_t)bh * 64 * 2048;

  // cooperative tile staging: 256 threads x 16B for K (32x64h) and V (64x32h)
  const int krow = tid >> 3, kchunk = tid & 7;
  const int vrow = tid >> 2, vchunk = tid & 3;
  auto issue_tile = [&](int kt, int buf) {
    async_copy16(&Ks[buf][krow * 64 + kchunk * 8],
                 Kbase + (size_t)(kt + krow) * 64 + kchunk * 8);
    async_copy16(&Vs[buf][vrow * 32 + vchunk * 8],
                 Vbase + (size_t)vrow * 2048 + kt + vchunk * 8);
  };

  issue_tile(0, 0);

  const half_t* Qbase = qb + (((size_t)bh * 1024) + q0) * 64;
  v16h qa0 = load_a_frag(Qbase + 0,  64, lane16, lh);
  v16h qa1 = load_a_frag(Qbase + 32, 64, lane16, lh);

  v8f z = {};
  v8f oacc[4];
  float rmax[8], rsum[8];
#pragma unroll
  for (int t = 0; t < 4; ++t) oacc[t] = z;
#pragma unroll
  for (int j = 0; j < 8; ++j) { rmax[j] = -3.0e38f; rsum[j] = 0.0f; }

  for (int it = 0; it < 64; ++it) {
    const int kt  = it * 32;
    const int buf = it & 1;
    if (it + 1 < 64) {
      issue_tile(kt + 32, (it + 1) & 1);
      wait_async_pending2();     // own tile `it` has landed (in-order)
    } else {
      wait_async_all();
    }
    __syncthreads();             // everyone's tile `it` visible

    v8f sacc[2];
#pragma unroll
    for (int sub = 0; sub < 2; ++sub) {
      sacc[sub] = z;
      const half_t* Kr = &Ks[buf][(sub * 16 + lane16) * 64 + lh * 16];
      v16h kb0 = *(const v16h*)(Kr);
      v16h kb1 = *(const v16h*)(Kr + 32);
      sacc[sub] = WMMA_F32_F16(qa0, kb0, sacc[sub]);
      sacc[sub] = WMMA_F32_F16(qa1, kb1, sacc[sub]);
    }
    // mask + online softmax; C-tile row = 8*lh + j, col = key lane16
#pragma unroll
    for (int j = 0; j < 8; ++j) {
      int qg = q0 + lh * 8 + j;
      int key0 = kt + lane16;
      int key1 = key0 + 16;
      bool v0 = (key0 < 1024) ? (key0 <= qg) : ((key0 - 1024) >= qg);
      bool v1 = (key1 < 1024) ? (key1 <= qg) : ((key1 - 1024) >= qg);
      float s0 = v0 ? (float)sacc[0][j] : -3.0e38f;
      float s1 = v1 ? (float)sacc[1][j] : -3.0e38f;
      float tm = fmaxf(s0, s1);
#pragma unroll
      for (int d = 1; d < 16; d <<= 1) tm = fmaxf(tm, __shfl_xor(tm, d, 32));
      float nm   = fmaxf(rmax[j], tm);
      float corr = __expf(rmax[j] - nm);
      rmax[j] = nm;
      float p0 = __expf(s0 - nm);
      float p1 = __expf(s1 - nm);
      psh[w][(lh * 8 + j) * 32 + lane16]      = (half_t)p0;
      psh[w][(lh * 8 + j) * 32 + 16 + lane16] = (half_t)p1;
      float ts = p0 + p1;
#pragma unroll
      for (int d = 1; d < 16; d <<= 1) ts += __shfl_xor(ts, d, 32);
      rsum[j] = rsum[j] * corr + ts;
#pragma unroll
      for (int t = 0; t < 4; ++t) oacc[t][j] *= corr;
    }
    // re-read P (16x32) from LDS in A-fragment layout
    const half_t* pp = &psh[w][lane16 * 32 + lh * 16];
    v16h pa = pack16(*(const v8h*)pp, *(const v8h*)(pp + 8));
#pragma unroll
    for (int t = 0; t < 4; ++t) {
      const half_t* Vr = &Vs[buf][(t * 16 + lane16) * 32 + lh * 16];
      v16h vb = *(const v16h*)Vr;
      oacc[t] = WMMA_F32_F16(pa, vb, oacc[t]);
    }
    __syncthreads();             // buffer `buf` free for reuse at it+2
  }
#pragma unroll
  for (int t = 0; t < 4; ++t)
#pragma unroll
    for (int j = 0; j < 8; ++j) {
      float val = (float)oacc[t][j] / rsum[j];
      int s = q0 + lh * 8 + j;
      int c = h * 64 + t * 16 + lane16;
      ob[((size_t)b * 1024 + s) * 1024 + c] = (half_t)val;
    }
}

// ---------------------------------------------------------------------------
// Output projection: x = comb + (o @ Wo + bo)  (fp32)
// ---------------------------------------------------------------------------
__global__ void k_gemm_o(const half_t* __restrict__ A, const half_t* __restrict__ Wt,
                         const float* __restrict__ bias, const float* __restrict__ comb,
                         float* __restrict__ x) {
  gemm_tile<1024>(A, Wt, [&](int m, int n, float v) {
    size_t idx = (size_t)m * 1024 + n;
    x[idx] = v + bias[n] + comb[idx];
  });
}

// ---------------------------------------------------------------------------
// h = LN_mlp(LN_2(x)) -> f16
// ---------------------------------------------------------------------------
__global__ void k_dln(const float* __restrict__ x, const float* __restrict__ g2,
                      const float* __restrict__ bb2, const float* __restrict__ gm,
                      const float* __restrict__ bm, half_t* __restrict__ h) {
  const int row = blockIdx.x;
  const int c0  = threadIdx.x * 4;
  const float* px = x + (size_t)row * 1024 + c0;
  float v[4];
#pragma unroll
  for (int i = 0; i < 4; ++i) v[i] = px[i];
  float s1 = 0.f, s2 = 0.f;
#pragma unroll
  for (int i = 0; i < 4; ++i) { s1 += v[i]; s2 += v[i] * v[i]; }
  float2 r = block_sum2(s1, s2);
  float mean = r.x * (1.0f / 1024.0f);
  float var  = r.y * (1.0f / 1024.0f) - mean * mean;
  float rstd = rsqrtf(var + 1e-5f);
  float y[4];
  s1 = 0.f; s2 = 0.f;
#pragma unroll
  for (int i = 0; i < 4; ++i) {
    y[i] = (v[i] - mean) * rstd * g2[c0 + i] + bb2[c0 + i];
    s1 += y[i]; s2 += y[i] * y[i];
  }
  r = block_sum2(s1, s2);
  mean = r.x * (1.0f / 1024.0f);
  var  = r.y * (1.0f / 1024.0f) - mean * mean;
  rstd = rsqrtf(var + 1e-5f);
  half_t* po = h + (size_t)row * 1024 + c0;
#pragma unroll
  for (int i = 0; i < 4; ++i)
    po[i] = (half_t)((y[i] - mean) * rstd * gm[c0 + i] + bm[c0 + i]);
}

// ---------------------------------------------------------------------------
// MLP: a1 = gelu_exact(h @ W1 + b1);  out = x + (a1 @ W2 + b2)
// ---------------------------------------------------------------------------
__global__ void k_gemm_m1(const half_t* __restrict__ A, const half_t* __restrict__ Wt,
                          const float* __restrict__ bias, half_t* __restrict__ a1) {
  gemm_tile<1024>(A, Wt, [&](int m, int n, float v) {
    float t = v + bias[n];
    float g = 0.5f * t * (1.0f + erff(t * 0.70710678118654752f));
    a1[(size_t)m * 4096 + n] = (half_t)g;
  });
}

__global__ void k_gemm_m2(const half_t* __restrict__ A, const half_t* __restrict__ Wt,
                          const float* __restrict__ bias, const float* __restrict__ x,
                          float* __restrict__ out) {
  gemm_tile<4096>(A, Wt, [&](int m, int n, float v) {
    size_t idx = (size_t)m * 1024 + n;
    out[idx] = v + bias[n] + x[idx];
  });
}

// ---------------------------------------------------------------------------
extern "C" void kernel_launch(void* const* d_in, const int* in_sizes, int n_in,
                              void* d_out, int out_size, void* d_ws, size_t ws_size,
                              hipStream_t stream) {
  (void)in_sizes; (void)n_in; (void)out_size; (void)ws_size;
  const float* fwd  = (const float*)d_in[0];
  const float* bwd  = (const float*)d_in[1];
  const float* pos  = (const float*)d_in[2];
  const float* ln1g = (const float*)d_in[3];
  const float* ln1b = (const float*)d_in[4];
  const float* Wq   = (const float*)d_in[5];
  const float* bq   = (const float*)d_in[6];
  const float* Wk   = (const float*)d_in[7];
  const float* bk   = (const float*)d_in[8];
  const float* Wv   = (const float*)d_in[9];
  const float* bv   = (const float*)d_in[10];
  const float* Wo   = (const float*)d_in[11];
  const float* bo   = (const float*)d_in[12];
  const float* ln2g = (const float*)d_in[13];
  const float* ln2b = (const float*)d_in[14];
  const float* mlpg = (const float*)d_in[15];
  const float* mlpb = (const float*)d_in[16];
  const float* W1   = (const float*)d_in[17];
  const float* b1   = (const float*)d_in[18];
  const float* W2   = (const float*)d_in[19];
  const float* b2   = (const float*)d_in[20];

  uint8_t* ws = (uint8_t*)d_ws;
  size_t off = 0;
  auto alloc = [&](size_t bytes) -> void* {
    void* p = ws + off;
    off += (bytes + 255) & ~(size_t)255;
    return p;
  };
  half_t* cx   = (half_t*)alloc(2048ull * 1024 * 2);
  half_t* ae   = (half_t*)alloc(4096ull * 1024 * 2);
  float*  comb = (float*) alloc(2048ull * 1024 * 4);
  half_t* Wqt  = (half_t*)alloc(1024ull * 1024 * 2);
  half_t* Wkt  = (half_t*)alloc(1024ull * 1024 * 2);
  half_t* Wvt  = (half_t*)alloc(1024ull * 1024 * 2);
  half_t* Wot  = (half_t*)alloc(1024ull * 1024 * 2);
  half_t* W1t  = (half_t*)alloc(4096ull * 1024 * 2);
  half_t* W2t  = (half_t*)alloc(1024ull * 4096 * 2);
  half_t* qbuf = (half_t*)alloc(2048ull * 1024 * 2);   // [B,H,S,64]
  half_t* kbuf = (half_t*)alloc(4096ull * 1024 * 2);   // [B,H,2S,64]
  half_t* vtb  = (half_t*)alloc(4096ull * 1024 * 2);   // [B,H,64,2S]
  half_t* obuf = (half_t*)alloc(2048ull * 1024 * 2);   // [B,S,C]
  float*  xbuf = (float*) alloc(2048ull * 1024 * 4);
  half_t* hbuf = (half_t*)alloc(2048ull * 1024 * 2);
  half_t* a1   = (half_t*)alloc(2048ull * 4096 * 2);

  // weight conversion + transpose (fp32 [K,N] -> f16 [N,K])
  k_trans<<<4096, 256, 0, stream>>>(Wq, Wqt, 1024, 1024);
  k_trans<<<4096, 256, 0, stream>>>(Wk, Wkt, 1024, 1024);
  k_trans<<<4096, 256, 0, stream>>>(Wv, Wvt, 1024, 1024);
  k_trans<<<4096, 256, 0, stream>>>(Wo, Wot, 1024, 1024);
  k_trans<<<16384, 256, 0, stream>>>(W1, W1t, 1024, 4096);
  k_trans<<<16384, 256, 0, stream>>>(W2, W2t, 4096, 1024);

  // comb + LayerNorms
  k_prep<<<6144, 256, 0, stream>>>(fwd, bwd, ln1g, ln1b, comb, cx, ae);

  // QKV projections (WMMA)
  k_gemm_q<<<dim3(8, 16), 256, 0, stream>>>(cx, Wqt, bq, pos, qbuf);
  k_gemm_k<<<dim3(8, 32), 256, 0, stream>>>(ae, Wkt, bk, pos, kbuf);
  k_gemm_v<<<dim3(8, 32), 256, 0, stream>>>(ae, Wvt, bv, vtb);

  // masked flash attention (WMMA + async LDS staging)
  k_attn<<<256, 256, 0, stream>>>(qbuf, kbuf, vtb, obuf);

  // output projection + residual
  k_gemm_o<<<dim3(8, 16), 256, 0, stream>>>(obuf, Wot, bo, comb, xbuf);

  // double LayerNorm
  k_dln<<<2048, 256, 0, stream>>>(xbuf, ln2g, ln2b, mlpg, mlpb, hbuf);

  // MLP (WMMA) + final residual into d_out
  k_gemm_m1<<<dim3(32, 16), 256, 0, stream>>>(hbuf, W1t, b1, a1);
  k_gemm_m2<<<dim3(8, 16), 256, 0, stream>>>(a1, W2t, b2, xbuf, (float*)d_out);
}
